// EdgeNetwork_57234734186739
// MI455X (gfx1250) — matrix operand order repaired and checked by reading, not compiled, not run on hardware
//
#include <hip/hip_runtime.h>
#include <hip/hip_bf16.h>

// ---------------------------------------------------------------------------
// EdgeNetwork: per-edge gather(concat dst,src) -> 256x256 -> 256x256 -> ReLU
// -> 256x1.  Tiled as GEMMs over 32-edge blocks using v_wmma_f32_16x16x32_f16
// (f16 operands, f32 accumulate) on gfx1250 wave32.
// K-step is the outer loop so each A-fragment is loaded from LDS once and
// feeds 4 WMMAs (4 column tiles held in live accumulators).
// ---------------------------------------------------------------------------

typedef _Float16 h16 __attribute__((ext_vector_type(16)));  // 8 VGPRs: WMMA A/B frag
typedef float    f8  __attribute__((ext_vector_type(8)));   // 8 VGPRs: WMMA C/D frag

#define E_PER_BLK 32
#define KDIM      256
#define LDA       264   // padded LDS row stride in f16 (264*2 = 528 B, 16B-aligned)

union Frag { h16 v; int4 q[2]; };

// One-time weight prep: W0T/W1T = f16 transpose ([n][k], K-contiguous rows),
// W2f = f16 copy of the 256-vector.
__global__ __launch_bounds__(256)
void prep_weights_kernel(const float* __restrict__ W0,
                         const float* __restrict__ W1,
                         const float* __restrict__ W2,
                         _Float16* __restrict__ W0T,
                         _Float16* __restrict__ W1T,
                         _Float16* __restrict__ W2f) {
    const int b = blockIdx.x;
    const int t = threadIdx.x;
    if (b < 256) {
        W0T[b * 256 + t] = (_Float16)W0[t * 256 + b];
    } else if (b < 512) {
        const int n = b - 256;
        W1T[n * 256 + t] = (_Float16)W1[t * 256 + n];
    } else {
        W2f[t] = (_Float16)W2[t];
    }
}

__global__ __launch_bounds__(256)
void edge_mlp_kernel(const float* __restrict__ node_h,
                     const int*   __restrict__ src_idx,
                     const int*   __restrict__ dst_idx,
                     const _Float16* __restrict__ W0T,
                     const float* __restrict__ b0,
                     const _Float16* __restrict__ W1T,
                     const float* __restrict__ b1,
                     const _Float16* __restrict__ W2f,
                     const float* __restrict__ b2,
                     float* __restrict__ out,
                     int n_edges) {
    __shared__ _Float16 Atile[E_PER_BLK * LDA];  // layer input tile (f16)
    __shared__ _Float16 Htile[E_PER_BLK * LDA];  // layer output tile (f16)

    const int tid  = threadIdx.x;
    const int wave = tid >> 5;       // 0..7
    const int lane = tid & 31;       // wave32 lane
    const int e0   = blockIdx.x * E_PER_BLK;

    // -------- Gather: 32 edges x 256 f16 features into LDS ------------------
    {
        const int e  = tid >> 3;                 // edge within block, 0..31
        const int c  = tid & 7;                  // 32-feature chunk, 0..7
        int eg = e0 + e;
        if (eg >= n_edges) eg = n_edges - 1;     // clamp (no divergence)
        const int node = (c < 4) ? dst_idx[eg] : src_idx[eg];
        const float4* __restrict__ p =
            (const float4*)(node_h + (size_t)node * 128 + (size_t)(c & 3) * 32);
        _Float16* dp = &Atile[e * LDA + c * 32];
#pragma unroll
        for (int j = 0; j < 4; ++j) {
            const float4 a = p[2 * j];
            const float4 b = p[2 * j + 1];
            union { _Float16 h[8]; int4 q; } u;
            u.h[0] = (_Float16)a.x; u.h[1] = (_Float16)a.y;
            u.h[2] = (_Float16)a.z; u.h[3] = (_Float16)a.w;
            u.h[4] = (_Float16)b.x; u.h[5] = (_Float16)b.y;
            u.h[6] = (_Float16)b.z; u.h[7] = (_Float16)b.w;
            *(int4*)(dp + j * 8) = u.q;
        }
    }
    __syncthreads();

    // Wave tiling: 8 waves = 2 row-blocks x 4 column groups (4x16 cols each).
    const int rb   = wave >> 2;      // row block (16 edges)
    const int cg   = wave & 3;       // column group (64 cols)
    const int m    = lane & 15;      // A-frag row within tile
    const int kh   = lane >> 4;      // K half select (ISA 16-bit frag layout)
    const int nsub = lane & 15;      // C/D column within tile
    const int nbase = cg * 64 + nsub; // column of tile t = nbase + 16*t

    // Prefetch layer-1 weights while layer 0 runs (global_prefetch_b8).
    __builtin_prefetch((const void*)(W1T + (size_t)nbase * KDIM), 0, 1);

    // -------- Layer 0: H = A @ W0 + b0 -------------------------------------
    {
        const _Float16* Arow = &Atile[(rb * 16 + m) * LDA];
        f8 acc[4] = {{0.f}, {0.f}, {0.f}, {0.f}};
#pragma unroll
        for (int kb = 0; kb < KDIM; kb += 32) {
            Frag af;
            // A 16x32 f16: lanes0-15 K=kb..+7 / kb+16..+23; lanes16-31 +8
            af.q[0] = *(const int4*)(Arow + kb + kh * 8);
            af.q[1] = *(const int4*)(Arow + kb + 16 + kh * 8);
#pragma unroll
            for (int t = 0; t < 4; ++t) {
                const _Float16* Bcol = &W0T[(size_t)(nbase + 16 * t) * KDIM];
                Frag bf;
                // B 32x16 f16: lanes0-15 K=kb..+7 / kb+16..+23; lanes16-31 +8
                bf.q[0] = *(const int4*)(Bcol + kb + kh * 8);
                bf.q[1] = *(const int4*)(Bcol + kb + 16 + kh * 8);
                acc[t] = __builtin_amdgcn_wmma_f32_16x16x32_f16(
                             false, af.v, false, bf.v, (short)0, acc[t],
                             false, false);
            }
        }
#pragma unroll
        for (int t = 0; t < 4; ++t) {
            const int n = nbase + 16 * t;
            const float bias = b0[n];
#pragma unroll
            for (int r = 0; r < 8; ++r)   // C/D: lane=N, row = r + 8*kh
                Htile[(rb * 16 + r + 8 * kh) * LDA + n] =
                    (_Float16)(acc[t][r] + bias);
        }
    }
    __syncthreads();

    __builtin_prefetch((const void*)W2f, 0, 1);

    // -------- Layer 1: A = relu(H @ W1 + b1) -------------------------------
    {
        const _Float16* Hrow = &Htile[(rb * 16 + m) * LDA];
        f8 acc[4] = {{0.f}, {0.f}, {0.f}, {0.f}};
#pragma unroll
        for (int kb = 0; kb < KDIM; kb += 32) {
            Frag af;
            af.q[0] = *(const int4*)(Hrow + kb + kh * 8);
            af.q[1] = *(const int4*)(Hrow + kb + 16 + kh * 8);
#pragma unroll
            for (int t = 0; t < 4; ++t) {
                const _Float16* Bcol = &W1T[(size_t)(nbase + 16 * t) * KDIM];
                Frag bf;
                bf.q[0] = *(const int4*)(Bcol + kb + kh * 8);
                bf.q[1] = *(const int4*)(Bcol + kb + 16 + kh * 8);
                acc[t] = __builtin_amdgcn_wmma_f32_16x16x32_f16(
                             false, af.v, false, bf.v, (short)0, acc[t],
                             false, false);
            }
        }
#pragma unroll
        for (int t = 0; t < 4; ++t) {
            const int n = nbase + 16 * t;
            const float bias = b1[n];
#pragma unroll
            for (int r = 0; r < 8; ++r) {
                float v = acc[t][r] + bias;
                v = v > 0.0f ? v : 0.0f;  // ReLU
                Atile[(rb * 16 + r + 8 * kh) * LDA + n] = (_Float16)v;
            }
        }
    }
    __syncthreads();

    // -------- Layer 2: out = A @ W2 + b2  (256-dot, 8 lanes per edge) ------
    {
        const int e  = tid >> 3;               // 0..31
        const int ks = (tid & 7) * 32;
        const _Float16* row = &Atile[e * LDA + ks];
        float s = 0.0f;
#pragma unroll
        for (int j = 0; j < 32; ++j)
            s += (float)row[j] * (float)W2f[ks + j];
        s += __shfl_xor(s, 1, 32);
        s += __shfl_xor(s, 2, 32);
        s += __shfl_xor(s, 4, 32);
        if ((tid & 7) == 0) {
            const int eg = e0 + e;
            if (eg < n_edges) out[eg] = s + b2[0];
        }
    }
}

extern "C" void kernel_launch(void* const* d_in, const int* in_sizes, int n_in,
                              void* d_out, int out_size, void* d_ws, size_t ws_size,
                              hipStream_t stream) {
    const float* node_h  = (const float*)d_in[0];
    const int*   src_idx = (const int*)d_in[1];
    const int*   dst_idx = (const int*)d_in[2];
    const float* W0      = (const float*)d_in[3];
    const float* b0      = (const float*)d_in[4];
    const float* W1      = (const float*)d_in[5];
    const float* b1      = (const float*)d_in[6];
    const float* W2      = (const float*)d_in[7];
    const float* b2      = (const float*)d_in[8];
    float* out = (float*)d_out;
    const int E = in_sizes[1];   // n_edges (src_idx length)

    // Workspace: f16 transposed weights (128KB + 128KB + 512B).
    _Float16* W0T = (_Float16*)d_ws;
    _Float16* W1T = W0T + 256 * 256;
    _Float16* W2f = W1T + 256 * 256;

    prep_weights_kernel<<<513, 256, 0, stream>>>(W0, W1, W2, W0T, W1T, W2f);

    const int blocks = (E + E_PER_BLK - 1) / E_PER_BLK;
    edge_mlp_kernel<<<blocks, 256, 0, stream>>>(node_h, src_idx, dst_idx,
                                                W0T, b0, W1T, b1, W2f, b2,
                                                out, E);
}
